// Multi_head_attention_57921928954349
// MI455X (gfx1250) — matrix operand order repaired and checked by reading, not compiled, not run on hardware
//
#include <hip/hip_runtime.h>
#include <hip/hip_bf16.h>
#include <stdint.h>

// ---------------------------------------------------------------------------
// CDNA5 (gfx1250) wave32 WMMA multi-head attention.
//   B=4, T=2048, E=1024, H=16, d=64
// ---------------------------------------------------------------------------

typedef __attribute__((ext_vector_type(16))) __bf16 v16bf;
typedef __attribute__((ext_vector_type(8)))  float  v8f;

union Frag16 {
  v16bf v;
  uint4 q[2];
};

static __device__ __forceinline__ v8f wmma_bf16(v16bf a, v16bf b, v8f c) {
  return __builtin_amdgcn_wmma_f32_16x16x32_bf16(false, a, false, b, (short)0, c,
                                                 false, false);
}

// ---------------------------------------------------------------------------
// f32 -> bf16 conversion (grid-stride)
// ---------------------------------------------------------------------------
__global__ void mha_cvt_bf16(const float* __restrict__ in,
                             __bf16* __restrict__ out, int n) {
  int i = blockIdx.x * blockDim.x + threadIdx.x;
  int stride = gridDim.x * blockDim.x;
  for (; i < n; i += stride) out[i] = (__bf16)in[i];
}

// ---------------------------------------------------------------------------
// GEMM: C[M,N] = A[M,K] (bf16, row-major) * W[N,K]^T (bf16, row-major, nn.Linear)
// Block: 256 threads = 8 waves, tile 128x128, K-step 32.
// Staging uses CDNA5 async global->LDS DMA (ASYNCcnt) with LDS double
// buffering: tile k+1 streams into the other buffer while WMMAs consume
// tile k. No VGPR staging, no loadcnt stalls in the hot loop.
// mode 0: outQKV bf16; z=0/1 -> [B,H,T,d] (Q,K); z=2 -> V TRANSPOSED [B,H,d,T]
//         so the attention context-WMMA B-fragments are contiguous.
// mode 1: outF f32 = C + bias[n], row-major [M,N].
// ---------------------------------------------------------------------------
#define GTM 128
#define GTN 128
#define GKT 32

__global__ __launch_bounds__(256) void mha_gemm_bf16(
    const __bf16* __restrict__ A, const __bf16* __restrict__ Wbase,
    __bf16* __restrict__ outQKV, float* __restrict__ outF,
    const float* __restrict__ bias, int M, int N, int K, int mode) {
  __shared__ alignas(16) __bf16 lA[2][GTM * GKT];
  __shared__ alignas(16) __bf16 lW[2][GTN * GKT];

  const int tid  = threadIdx.x;
  const int wave = tid >> 5;
  const int lane = tid & 31;
  const int half = lane >> 4;
  const int l16  = lane & 15;
  const int wm   = wave & 3;   // 0..3 : 32-row group
  const int wn   = wave >> 2;  // 0..1 : 64-col group
  const int mBase = blockIdx.x * GTM;
  const int nBase = blockIdx.y * GTN;
  const __bf16* W = Wbase + (size_t)blockIdx.z * (size_t)N * (size_t)K;

  v8f acc[2][4];
#pragma unroll
  for (int i = 0; i < 2; ++i)
#pragma unroll
    for (int j = 0; j < 4; ++j)
      acc[i][j] = (v8f){0.f, 0.f, 0.f, 0.f, 0.f, 0.f, 0.f, 0.f};

  const int ldr = tid >> 1;          // 0..127 : tile row
  const int ldk = (tid & 1) * 16;    // 0 or 16 within K-step

  // Async DMA of one 128x32 bf16 tile pair into LDS buffer `buf`.
  // Each thread copies 32 B of A and 32 B of W (two b128 ops each; the
  // instruction offset advances both the LDS and memory address).
  auto stage_async = [&](int buf, int k0) {
    unsigned ldsA = (unsigned)(uintptr_t)(&lA[buf][ldr * GKT + ldk]);
    unsigned long long gA =
        (unsigned long long)(uintptr_t)(A + (size_t)(mBase + ldr) * K + k0 + ldk);
    unsigned ldsW = (unsigned)(uintptr_t)(&lW[buf][ldr * GKT + ldk]);
    unsigned long long gW =
        (unsigned long long)(uintptr_t)(W + (size_t)(nBase + ldr) * K + k0 + ldk);
    asm volatile(
        "global_load_async_to_lds_b128 %0, %1, off\n\t"
        "global_load_async_to_lds_b128 %0, %1, off offset:16\n\t"
        "global_load_async_to_lds_b128 %2, %3, off\n\t"
        "global_load_async_to_lds_b128 %2, %3, off offset:16"
        :
        : "v"(ldsA), "v"(gA), "v"(ldsW), "v"(gW)
        : "memory");
  };

  const int nit = K / GKT;
  stage_async(0, 0);

  for (int it = 0; it < nit; ++it) {
    const int cur = it & 1;
    // Only this iteration's 4 async copies are outstanding here.
    asm volatile("s_wait_asynccnt 0x0" ::: "memory");
    __syncthreads();  // tile `cur` visible to all waves
    if (it + 1 < nit) stage_async(1 - cur, (it + 1) * GKT);  // prefetch

    // A fragments (16x32 bf16 layout: lanes 0-15 rows, K 0..7 & 16..23;
    // lanes 16-31 same rows, K 8..15 & 24..31).
    Frag16 af[2];
#pragma unroll
    for (int i = 0; i < 2; ++i) {
      const __bf16* base = &lA[cur][(wm * 32 + i * 16 + l16) * GKT + half * 8];
      af[i].q[0] = *(const uint4*)(base);
      af[i].q[1] = *(const uint4*)(base + 16);
    }
    // B fragments (32x16: lane n holds column n; lanes 0-15 K 0..7 & 16..23,
    // lanes 16-31 K 8..15 & 24..31). B[k][n] = W[n][k] -> contiguous in W row.
    Frag16 bfr[4];
#pragma unroll
    for (int j = 0; j < 4; ++j) {
      const __bf16* base = &lW[cur][(wn * 64 + j * 16 + l16) * GKT + half * 8];
      bfr[j].q[0] = *(const uint4*)(base);
      bfr[j].q[1] = *(const uint4*)(base + 16);
    }
#pragma unroll
    for (int i = 0; i < 2; ++i)
#pragma unroll
      for (int j = 0; j < 4; ++j)
        acc[i][j] = wmma_bf16(af[i].v, bfr[j].v, acc[i][j]);

    __syncthreads();  // all waves done reading `cur` before it is re-DMA'd
  }

  // Epilogue. C/D layout: VGPR e, lane -> M = e + 8*(lane>>4), N = lane&15.
#pragma unroll
  for (int i = 0; i < 2; ++i)
#pragma unroll
    for (int j = 0; j < 4; ++j)
#pragma unroll
      for (int e = 0; e < 8; ++e) {
        const int m = mBase + wm * 32 + i * 16 + e + 8 * half;
        const int n = nBase + wn * 64 + j * 16 + l16;
        const float v = acc[i][j][e];
        if (mode == 0) {
          const int b = m >> 11, t = m & 2047;
          const int h = n >> 6, d = n & 63;
          size_t idx;
          if (blockIdx.z == 2) {
            // V stored transposed: [B,H,d,T] (L2 absorbs the 2B scatter).
            idx = ((size_t)(b * 16 + h) * 64 + d) * 2048 + t;
          } else {
            // Q,K stored [B,H,T,d].
            idx = ((size_t)(b * 16 + h) * 2048 + t) * 64 + d;
          }
          outQKV[(size_t)blockIdx.z * 8388608ull + idx] = (__bf16)v;
        } else {
          outF[(size_t)m * N + n] = v + bias[n];
        }
      }
}

// ---------------------------------------------------------------------------
// Flash attention: per block (q-block of 64, head, batch), 128 threads =
// 4 fully independent waves (no block barriers); each wave owns a 16-row
// query tile and streams 32-key kv tiles. V arrives pre-transposed [B,H,d,T]
// so both K- and V-fragments are two contiguous 16B global loads per lane.
// ---------------------------------------------------------------------------
__global__ __launch_bounds__(128) void mha_flash_attn(
    const __bf16* __restrict__ Qm, const __bf16* __restrict__ Km,
    const __bf16* __restrict__ Vtm, const unsigned char* __restrict__ mask,
    __bf16* __restrict__ ctx) {
  __shared__ alignas(16) __bf16 lP[4][16 * 32];  // per-wave P tile [row][key]

  const int tid  = threadIdx.x;
  const int wave = tid >> 5;
  const int lane = tid & 31;
  const int half = lane >> 4;
  const int l16  = lane & 15;
  const int h = blockIdx.y, b = blockIdx.z;
  const size_t bh = ((size_t)(b * 16 + h)) * 2048ull * 64ull;
  const __bf16* Qp = Qm + bh;
  const __bf16* Kp = Km + bh;
  const __bf16* Vt = Vtm + bh;  // [d][T]: element (d,t) at d*2048 + t
  const unsigned char* mk = mask + (size_t)b * 2048;
  const int qt0 = blockIdx.x * 64 + wave * 16;

  // Q fragments (A-matrix 16x32, two K-chunks covering d=0..63), loaded once.
  Frag16 qf[2];
  {
    const int row = qt0 + l16;
#pragma unroll
    for (int f = 0; f < 2; ++f) {
      const __bf16* base = Qp + (size_t)row * 64 + f * 32 + half * 8;
      qf[f].q[0] = *(const uint4*)(base);
      qf[f].q[1] = *(const uint4*)(base + 16);
    }
  }

  v8f acc[4];
#pragma unroll
  for (int j = 0; j < 4; ++j)
    acc[j] = (v8f){0.f, 0.f, 0.f, 0.f, 0.f, 0.f, 0.f, 0.f};
  float mrow[8], lrow[8];
#pragma unroll
  for (int r = 0; r < 8; ++r) { mrow[r] = -1e30f; lrow[r] = 0.f; }

  for (int kt = 0; kt < 2048; kt += 32) {
    // --- Scores: S[q, key] over 32 keys = two 16x16 WMMA N-tiles, K-dim=64.
    v8f S[2];
#pragma unroll
    for (int nt = 0; nt < 2; ++nt) {
      const int key = kt + nt * 16 + l16;
      const __bf16* kb = Kp + (size_t)key * 64;
      Frag16 kf0, kf1;  // B-matrix: lane = key column; K(=d) runs in VGPRs
      kf0.q[0] = *(const uint4*)(kb + half * 8);
      kf0.q[1] = *(const uint4*)(kb + half * 8 + 16);
      kf1.q[0] = *(const uint4*)(kb + 32 + half * 8);
      kf1.q[1] = *(const uint4*)(kb + 32 + half * 8 + 16);
      v8f z = (v8f){0.f, 0.f, 0.f, 0.f, 0.f, 0.f, 0.f, 0.f};
      z = wmma_bf16(qf[0].v, kf0.v, z);
      z = wmma_bf16(qf[1].v, kf1.v, z);
      const bool masked = mk[key] != 0;  // per-lane key column
#pragma unroll
      for (int e = 0; e < 8; ++e) z[e] = masked ? -1e9f : z[e] * 0.125f;
      S[nt] = z;
    }

    // --- Online softmax. Row M = r + 8*half lives in VGPR r across the
    // 16-lane half; xor-shuffles 1/2/4/8 reduce within the half.
    float scl[8];
#pragma unroll
    for (int r = 0; r < 8; ++r) {
      float v = fmaxf(S[0][r], S[1][r]);
      v = fmaxf(v, __shfl_xor(v, 1));
      v = fmaxf(v, __shfl_xor(v, 2));
      v = fmaxf(v, __shfl_xor(v, 4));
      v = fmaxf(v, __shfl_xor(v, 8));
      const float nm = fmaxf(mrow[r], v);
      scl[r] = __expf(mrow[r] - nm);
      mrow[r] = nm;
      const float p0 = __expf(S[0][r] - nm);
      const float p1 = __expf(S[1][r] - nm);
      S[0][r] = p0;
      S[1][r] = p1;
      float s = p0 + p1;
      s += __shfl_xor(s, 1);
      s += __shfl_xor(s, 2);
      s += __shfl_xor(s, 4);
      s += __shfl_xor(s, 8);
      lrow[r] = lrow[r] * scl[r] + s;
    }
#pragma unroll
    for (int j = 0; j < 4; ++j)
#pragma unroll
      for (int r = 0; r < 8; ++r) acc[j][r] = acc[j][r] * scl[r];

    // --- P (C-layout) -> per-wave LDS bf16 [row][key], reload in A-layout.
    __bf16* pw = &lP[wave][0];
#pragma unroll
    for (int r = 0; r < 8; ++r) {
      const int row = r + 8 * half;
      pw[row * 32 + l16]      = (__bf16)S[0][r];
      pw[row * 32 + 16 + l16] = (__bf16)S[1][r];
    }
    Frag16 pf;
    {
      const __bf16* base = pw + l16 * 32 + half * 8;
      pf.q[0] = *(const uint4*)(base);
      pf.q[1] = *(const uint4*)(base + 16);
    }
    // --- Context: acc[j] += P[16x32keys] * V[32keys x 16d].
    // V^T[d][t]: lane = d column, lane half h holds keys 8h..8h+7 and 16+8h..23+8h.
#pragma unroll
    for (int j = 0; j < 4; ++j) {
      const __bf16* vb = Vt + (size_t)(j * 16 + l16) * 2048 + kt + half * 8;
      Frag16 vf;
      vf.q[0] = *(const uint4*)(vb);
      vf.q[1] = *(const uint4*)(vb + 16);
      acc[j] = wmma_bf16(pf.v, vf.v, acc[j]);
    }
  }

  // Epilogue: normalize and write context bf16 [B,T,E] (E = h*64 + d).
#pragma unroll
  for (int j = 0; j < 4; ++j)
#pragma unroll
    for (int r = 0; r < 8; ++r) {
      const int t = qt0 + r + 8 * half;
      const int d = j * 16 + l16;
      const float o = acc[j][r] / lrow[r];
      ctx[((size_t)b * 2048 + t) * 1024 + h * 64 + d] = (__bf16)o;
    }
}

// ---------------------------------------------------------------------------
// Host launcher
// ---------------------------------------------------------------------------
extern "C" void kernel_launch(void* const* d_in, const int* in_sizes, int n_in,
                              void* d_out, int out_size, void* d_ws,
                              size_t ws_size, hipStream_t stream) {
  const float* x          = (const float*)d_in[0];
  const unsigned char* pm = (const unsigned char*)d_in[1];  // bool [B,1,1,T]
  const float* Wq = (const float*)d_in[2];
  const float* Wk = (const float*)d_in[3];
  const float* Wv = (const float*)d_in[4];
  const float* Wo = (const float*)d_in[5];
  const float* bo = (const float*)d_in[6];
  float* out = (float*)d_out;

  const int M = 8192;            // B*T
  const int E = 1024;
  const size_t XW   = (size_t)M * E;        // 8,388,608 elems
  const size_t WW   = (size_t)E * E;        // 1,048,576 elems

  char* ws = (char*)d_ws;
  __bf16* xb  = (__bf16*)(ws);                              // 16 MiB
  __bf16* wb  = (__bf16*)(ws + XW * 2);                     //  8 MiB (Wq,Wk,Wv,Wo)
  __bf16* qkv = (__bf16*)(ws + XW * 2 + 4 * WW * 2);        // 48 MiB (Q,K,V^T)
  __bf16* ctx = (__bf16*)(ws + XW * 2 + 4 * WW * 2 + 3 * XW * 2);  // 16 MiB

  // 1) Convert activations and weights to bf16.
  mha_cvt_bf16<<<2048, 256, 0, stream>>>(x, xb, (int)XW);
  mha_cvt_bf16<<<1024, 256, 0, stream>>>(Wq, wb + 0 * WW, (int)WW);
  mha_cvt_bf16<<<1024, 256, 0, stream>>>(Wk, wb + 1 * WW, (int)WW);
  mha_cvt_bf16<<<1024, 256, 0, stream>>>(Wv, wb + 2 * WW, (int)WW);
  mha_cvt_bf16<<<1024, 256, 0, stream>>>(Wo, wb + 3 * WW, (int)WW);

  // 2) Fused Q/K/V projections (z selects weight); Q,K -> [B,H,T,d],
  //    V -> transposed [B,H,d,T].
  mha_gemm_bf16<<<dim3(M / GTM, E / GTN, 3), 256, 0, stream>>>(
      xb, wb, qkv, nullptr, nullptr, M, E, E, 0);

  // 3) Flash attention (barrier-free inner loop) -> ctx bf16 [B,T,E].
  mha_flash_attn<<<dim3(32, 16, 4), 128, 0, stream>>>(
      qkv, qkv + XW, qkv + 2 * XW, pm, ctx);

  // 4) Output projection + bias -> f32 d_out.
  mha_gemm_bf16<<<dim3(M / GTM, E / GTN, 1), 256, 0, stream>>>(
      ctx, wb + 3 * WW, nullptr, out, bo, M, E, E, 1);

  (void)in_sizes; (void)n_in; (void)out_size; (void)ws_size;
}
